// RobustConv_82377472737746
// MI455X (gfx1250) — matrix operand order repaired and checked by reading, not compiled, not run on hardware
//
#include <hip/hip_runtime.h>
#include <cstdint>

typedef __attribute__((ext_vector_type(16))) _Float16 v16h;
typedef __attribute__((ext_vector_type(8)))  float    v8f;

#define N_NODES 50000
#define N_EDGES 600000
#define IN_C    256
#define OUT_C   128

// ---------------------------------------------------------------- degree prep
__global__ void init_deg_kernel(float* __restrict__ dinv) {
  int i = blockIdx.x * blockDim.x + threadIdx.x;
  if (i < N_NODES) dinv[i] = 1.0f;  // self-loop contributes 1 to degree
}

__global__ void deg_kernel(const long long* __restrict__ ei, float* __restrict__ dinv) {
  int e = blockIdx.x * blockDim.x + threadIdx.x;
  if (e < N_EDGES) {
    int c = (int)ei[N_EDGES + e];   // col (source) row of edge_index
    atomicAdd(&dinv[c], 1.0f);
  }
}

__global__ void rsqrt_kernel(float* __restrict__ dinv) {
  int i = blockIdx.x * blockDim.x + threadIdx.x;
  if (i < N_NODES) dinv[i] = rsqrtf(dinv[i]);   // deg >= 1 always (self loop)
}

// ------------------------------------------------- W transpose + f16 convert
// wt[mat][n][k] = W_mat[k][n]  (k contiguous per output column)
__global__ void prep_w_kernel(const float* __restrict__ Wm,
                              const float* __restrict__ Wv,
                              _Float16* __restrict__ wt) {
  int t = blockIdx.x * blockDim.x + threadIdx.x;  // 2*128*256 = 65536
  if (t < 2 * OUT_C * IN_C) {
    int mat = t / (OUT_C * IN_C);
    int r   = t % (OUT_C * IN_C);
    int n   = r / IN_C;
    int k   = r % IN_C;
    const float* W = mat ? Wv : Wm;
    wt[t] = (_Float16)W[k * OUT_C + n];
  }
}

// --------------------------------------------------------------- WMMA GEMM
// Block = 8 waves. Wave w: rows rowTile..rowTile+15 (rowTile = blk*32 + (w>>2)*16),
// cols (w&3)*32..+31, computing BOTH mean and var tiles so the attention
// epilogue is wave-local. f16 A/B, f32 accumulate.
__global__ __launch_bounds__(256) void gemm_att_kernel(
    const float* __restrict__ x, const _Float16* __restrict__ wt,
    const float* __restrict__ b_mean, const float* __restrict__ b_var,
    float* __restrict__ mean_att, float* __restrict__ var_att) {
  const int tid  = threadIdx.x;
  const int wave = tid >> 5;
  const int lane = tid & 31;
  const int hi   = (lane >> 4) & 1;      // lane half: 0 = lanes 0-15, 1 = 16-31
  const int ln   = lane & 15;
  const int rowTile = blockIdx.x * 32 + (wave >> 2) * 16;
  const int colBase = (wave & 3) * 32;

  // A: one matrix row per lane (lanes 0-15 and 16-31 both cover rows 0-15)
  int aRow = rowTile + ln;
  if (aRow >= N_NODES) aRow = N_NODES - 1;           // clamp; stores predicated
  const float* xrow = x + (size_t)aRow * IN_C;

  // B: per-lane column pointers into transposed f16 weights
  const _Float16* wtm0 = wt + (size_t)(colBase + ln)              * IN_C;
  const _Float16* wtm1 = wt + (size_t)(colBase + 16 + ln)         * IN_C;
  const _Float16* wtv0 = wt + (size_t)(OUT_C + colBase + ln)      * IN_C;
  const _Float16* wtv1 = wt + (size_t)(OUT_C + colBase + 16 + ln) * IN_C;
  const int bksel = hi * 16;  // lanes 16-31 hold K=16..31 of the 32-K B tile

  v8f cm0 = {}, cm1 = {}, cv0 = {}, cv1 = {};

#pragma unroll
  for (int k0 = 0; k0 < IN_C; k0 += 32) {
    // A fragment, contiguous K layout (matches B):
    // halves 0-7  <- K = k0 + hi*16 + 0..7
    // halves 8-15 <- K = k0 + hi*16 + 8..15
    const int kb = k0 + hi * 16;
    float4 f0 = *(const float4*)(xrow + kb);
    float4 f1 = *(const float4*)(xrow + kb + 4);
    float4 f2 = *(const float4*)(xrow + kb + 8);
    float4 f3 = *(const float4*)(xrow + kb + 12);
    v16h a;
    a[0]=(_Float16)f0.x;  a[1]=(_Float16)f0.y;  a[2]=(_Float16)f0.z;  a[3]=(_Float16)f0.w;
    a[4]=(_Float16)f1.x;  a[5]=(_Float16)f1.y;  a[6]=(_Float16)f1.z;  a[7]=(_Float16)f1.w;
    a[8]=(_Float16)f2.x;  a[9]=(_Float16)f2.y;  a[10]=(_Float16)f2.z; a[11]=(_Float16)f2.w;
    a[12]=(_Float16)f3.x; a[13]=(_Float16)f3.y; a[14]=(_Float16)f3.z; a[15]=(_Float16)f3.w;

    v16h bm0 = *(const v16h*)(wtm0 + k0 + bksel);
    v16h bm1 = *(const v16h*)(wtm1 + k0 + bksel);
    v16h bv0 = *(const v16h*)(wtv0 + k0 + bksel);
    v16h bv1 = *(const v16h*)(wtv1 + k0 + bksel);

    cm0 = __builtin_amdgcn_wmma_f32_16x16x32_f16(false, a, false, bm0, (short)0, cm0, false, false);
    cm1 = __builtin_amdgcn_wmma_f32_16x16x32_f16(false, a, false, bm1, (short)0, cm1, false, false);
    cv0 = __builtin_amdgcn_wmma_f32_16x16x32_f16(false, a, false, bv0, (short)0, cv0, false, false);
    cv1 = __builtin_amdgcn_wmma_f32_16x16x32_f16(false, a, false, bv1, (short)0, cv1, false, false);
  }

  // Epilogue. C/D layout: element i -> row rowTile + i + hi*8, col = tileCol + ln.
  const int rbase = rowTile + hi * 8;
#pragma unroll
  for (int t = 0; t < 2; ++t) {
    const int n = colBase + t * 16 + ln;
    const float bm = b_mean[n];
    const float bv = b_var[n];
    const v8f cm = t ? cm1 : cm0;
    const v8f cv = t ? cv1 : cv0;
#pragma unroll
    for (int i = 0; i < 8; ++i) {
      const int row = rbase + i;
      if (row < N_NODES) {
        float m = cm[i] + bm; m = m > 0.0f ? m : 0.0f;   // relu
        float v = cv[i] + bv; v = v > 0.0f ? v : 0.0f;
        float att = __expf(-v);                          // GAMMA = 1
        mean_att[(size_t)row * OUT_C + n] = m * att;
        var_att[(size_t)row * OUT_C + n]  = v * att * att;
      }
    }
  }
}

// ------------------------------------------- self-loop init of the outputs
__global__ void self_loop_kernel(const float* __restrict__ dinv,
                                 const float* __restrict__ mean_att,
                                 const float* __restrict__ var_att,
                                 float* __restrict__ out_mean,
                                 float* __restrict__ out_var) {
  int t = blockIdx.x * blockDim.x + threadIdx.x;
  if (t < N_NODES * OUT_C) {
    int i = t >> 7;                    // node
    float di = dinv[i];
    float w  = di * di;                // self-loop weight = 1/deg
    out_mean[t] = w * mean_att[t];     // plain store: initializes d_out
    out_var[t]  = (w * w) * var_att[t];
  }
}

// --------------------------------------------------- edge scatter (SpMM)
__global__ __launch_bounds__(256) void scatter_kernel(
    const long long* __restrict__ ei, const float* __restrict__ dinv,
    const float* __restrict__ mean_att, const float* __restrict__ var_att,
    float* __restrict__ out_mean, float* __restrict__ out_var) {
  const long long t = (long long)blockIdx.x * blockDim.x + threadIdx.x;
  const int e = (int)(t >> 5);                 // 32 threads per edge
  if (e >= N_EDGES) return;
  const int cg = ((int)t & 31) * 4;            // 4 channels per thread
  const int r = (int)ei[e];                    // target
  const int c = (int)ei[N_EDGES + e];          // source
  const float w  = dinv[r] * dinv[c];
  const float w2 = w * w;
  const float4 m = *(const float4*)(mean_att + (size_t)c * OUT_C + cg);
  const float4 v = *(const float4*)(var_att  + (size_t)c * OUT_C + cg);
  float* om = out_mean + (size_t)r * OUT_C + cg;
  float* ov = out_var  + (size_t)r * OUT_C + cg;
  atomicAdd(om + 0, w  * m.x); atomicAdd(om + 1, w  * m.y);
  atomicAdd(om + 2, w  * m.z); atomicAdd(om + 3, w  * m.w);
  atomicAdd(ov + 0, w2 * v.x); atomicAdd(ov + 1, w2 * v.y);
  atomicAdd(ov + 2, w2 * v.z); atomicAdd(ov + 3, w2 * v.w);
}

// ----------------------------------------------------------------- launch
extern "C" void kernel_launch(void* const* d_in, const int* in_sizes, int n_in,
                              void* d_out, int out_size, void* d_ws, size_t ws_size,
                              hipStream_t stream) {
  (void)in_sizes; (void)n_in; (void)out_size; (void)ws_size;
  const float*     x      = (const float*)d_in[0];
  const long long* ei     = (const long long*)d_in[1];   // int64 edge_index [2, E]
  const float*     W_mean = (const float*)d_in[2];
  const float*     W_var  = (const float*)d_in[3];
  const float*     b_mean = (const float*)d_in[4];
  const float*     b_var  = (const float*)d_in[5];

  float* out_mean = (float*)d_out;
  float* out_var  = out_mean + (size_t)N_NODES * OUT_C;

  // workspace layout
  char* ws = (char*)d_ws;
  size_t off = 0;
  _Float16* wt = (_Float16*)(ws + off); off += (size_t)2 * OUT_C * IN_C * sizeof(_Float16);
  off = (off + 255) & ~(size_t)255;
  float* dinv = (float*)(ws + off);     off += (size_t)N_NODES * sizeof(float);
  off = (off + 255) & ~(size_t)255;
  float* mean_att = (float*)(ws + off); off += (size_t)N_NODES * OUT_C * sizeof(float);
  float* var_att  = (float*)(ws + off);

  const int T = 256;
  hipLaunchKernelGGL(prep_w_kernel, dim3((2 * OUT_C * IN_C + T - 1) / T), dim3(T), 0, stream,
                     W_mean, W_var, wt);
  hipLaunchKernelGGL(init_deg_kernel, dim3((N_NODES + T - 1) / T), dim3(T), 0, stream, dinv);
  hipLaunchKernelGGL(deg_kernel, dim3((N_EDGES + T - 1) / T), dim3(T), 0, stream, ei, dinv);
  hipLaunchKernelGGL(rsqrt_kernel, dim3((N_NODES + T - 1) / T), dim3(T), 0, stream, dinv);
  hipLaunchKernelGGL(gemm_att_kernel, dim3((N_NODES + 31) / 32), dim3(T), 0, stream,
                     x, wt, b_mean, b_var, mean_att, var_att);
  hipLaunchKernelGGL(self_loop_kernel, dim3((N_NODES * OUT_C + T - 1) / T), dim3(T), 0, stream,
                     dinv, mean_att, var_att, out_mean, out_var);
  hipLaunchKernelGGL(scatter_kernel,
                     dim3((int)(((long long)N_EDGES * 32 + T - 1) / T)), dim3(T), 0, stream,
                     ei, dinv, mean_att, var_att, out_mean, out_var);
}
